// MultiQuerySelfAttention_61280593379989
// MI455X (gfx1250) — matrix operand order repaired and checked by reading, not compiled, not run on hardware
//
#include <hip/hip_runtime.h>
#include <math.h>

// Problem constants (fixed by the reference).
#define B_   4
#define S_   2048
#define H_   2048
#define NH_  16
#define HD_  128
#define BS_  (B_*S_)

typedef __attribute__((ext_vector_type(16))) __bf16 v16bf;
typedef __attribute__((ext_vector_type(8)))  float  v8f;
typedef __attribute__((ext_vector_type(4))) unsigned int u32x4;
typedef __attribute__((ext_vector_type(8))) int          i32x8;
typedef __attribute__((ext_vector_type(4))) int          i32x4;

#if defined(__has_builtin)
#if __has_builtin(__builtin_amdgcn_tensor_load_to_lds)
#define HAVE_TDM 1
#endif
#if __has_builtin(__builtin_amdgcn_s_wait_tensorcnt)
#define HAVE_TCNT 1
#endif
#endif

union BF16x16 { v16bf v; uint4 q[2]; unsigned short u[16]; };

__device__ __forceinline__ unsigned short f2bf(float f) {
  unsigned u = __float_as_uint(f);
  u += 0x7FFFu + ((u >> 16) & 1u);              // round-to-nearest-even
  return (unsigned short)(u >> 16);
}
__device__ __forceinline__ float bf2f(unsigned short h) {
  return __uint_as_float(((unsigned)h) << 16);
}
__device__ __forceinline__ v8f v8f_zero() {
  v8f z;
#pragma unroll
  for (int i = 0; i < 8; ++i) z[i] = 0.f;
  return z;
}
__device__ __forceinline__ v8f wmma_bf16(v16bf a, v16bf b, v8f c) {
  return __builtin_amdgcn_wmma_f32_16x16x32_bf16(false, a, false, b, (short)0, c,
                                                 false, false);
}

// A-fragment (16x32 bf16, MxK) from a row-major tile (pitch in ushorts).
__device__ __forceinline__ v16bf fragA_mem(const unsigned short* t, int pitch, int lane) {
  BF16x16 r;
  const unsigned short* p = t + (lane & 15) * pitch + ((lane & 16) ? 16 : 0);
  r.q[0] = *(const uint4*)p;
  r.q[1] = *(const uint4*)(p + 8);
  return r.v;
}
// B-fragment (32x16 bf16, KxN) from a tile storing B^T row-major (row n holds
// 32 contiguous K values).  Lanes 0-15: n=lane, K 0..15; lanes 16-31: K 16..31.
__device__ __forceinline__ v16bf fragB_mem(const unsigned short* t, int pitch, int lane) {
  BF16x16 r;
  const uint4* p = (const uint4*)(t + (lane & 15) * pitch + ((lane & 16) ? 16 : 0));
  r.q[0] = p[0];
  r.q[1] = p[1];
  return r.v;
}

#if defined(HAVE_TDM)
// Issue one 2-D TENSOR_LOAD_TO_LDS: tile_rows x tile_cols elements of 2 bytes,
// source row pitch row_stride elements, packed row-major into LDS at lds_byte_off.
__device__ __forceinline__ void tdm_load_2d(unsigned lds_byte_off,
                                            const unsigned short* gsrc,
                                            unsigned tile_cols, unsigned tile_rows,
                                            unsigned row_stride) {
  unsigned long long ga = (unsigned long long)(const void*)gsrc;
  u32x4 g0 = { 1u,                                    // count=1
               lds_byte_off,                          // lds_addr
               (unsigned)(ga & 0xFFFFFFFFull),
               (unsigned)((ga >> 32) & 0x01FFFFFFull) | 0x80000000u };  // type=2
  i32x8 g1 = { (int)(1u << 16),                                       // data_size=2B
               (int)((tile_cols & 0xFFFFu) << 16),                    // tdim0 lo16
               (int)(((tile_cols >> 16) & 0xFFFFu) |
                     ((tile_rows & 0xFFFFu) << 16)),                  // tdim0 hi|tdim1 lo
               (int)(((tile_rows >> 16) & 0xFFFFu) |
                     ((tile_cols & 0xFFFFu) << 16)),                  // tdim1 hi|tile0
               (int)(tile_rows & 0xFFFFu),                            // tile1 | tile2=0
               (int)row_stride,                                       // stride0 lo32
               0, 0 };
  i32x4 g2 = { 0, 0, 0, 0 };
  i32x4 g3 = { 0, 0, 0, 0 };
  i32x8 g4 = { 0, 0, 0, 0, 0, 0, 0, 0 };
  __builtin_amdgcn_tensor_load_to_lds(g0, g1, g2, g3, g4, 0);
}
#endif

__device__ __forceinline__ void wait_tensor0() {
#if defined(HAVE_TCNT)
  __builtin_amdgcn_s_wait_tensorcnt(0);
#else
  asm volatile("s_wait_tensorcnt 0x0" ::: "memory");
#endif
}

// ---------------------------------------------------------------------------
// fp32 -> bf16 conversion (x and the four weight matrices), float4-vectorized.
// ---------------------------------------------------------------------------
__global__ void cvt_bf16_kernel(const float* __restrict__ src,
                                unsigned short* __restrict__ dst) {
  int i = blockIdx.x * 256 + threadIdx.x;
  float4 v = ((const float4*)src)[i];
  *(uint2*)(dst + (size_t)i * 4) =
      make_uint2(f2bf(v.x) | ((unsigned)f2bf(v.y) << 16),
                 f2bf(v.z) | ((unsigned)f2bf(v.w) << 16));
}

// ---------------------------------------------------------------------------
// GEMM:  C[M=BS, N=H] = A[M,K=H] * W[N,K]^T, all-bf16 operands.
// 128x128x32 block tile, 8 waves. Both LDS tiles are staged by the Tensor
// Data Mover (double-buffered): wave 0 issues the descriptor pair for the
// next k-step, the DMA runs underneath the current step's 8 WMMAs, and one
// barrier per k-step orders TDM-write -> ds-read and ds-read -> TDM-write.
// OUT_MODE: 0 fp32 row-major; 1 bf16 [B,NH,S,HD]; 2 bf16 [B,NH,HD,S] (V^T).
// ---------------------------------------------------------------------------
template <int OUT_MODE>
__global__ __launch_bounds__(256) void gemm_bf16(const unsigned short* __restrict__ A,
                                                 const unsigned short* __restrict__ Wb,
                                                 void* __restrict__ Dst) {
  // SMEM (ushort): As0 @0, As1 @4096, Bs0 @8192, Bs1 @12288  (byte offs x2)
  __shared__ __align__(16) unsigned short SMEM[4 * 4096];
  const int tid = threadIdx.x, lane = tid & 31, w = tid >> 5;
  const int m0b = blockIdx.x * 128, n0b = blockIdx.y * 128;
  const int wm = (w >> 2) * 64, wn = (w & 3) * 32;

  v8f acc[4][2];
#pragma unroll
  for (int i = 0; i < 4; ++i)
#pragma unroll
    for (int j = 0; j < 2; ++j) acc[i][j] = v8f_zero();

#if defined(HAVE_TDM)
  if (w == 0) {  // prologue: stage tile 0 into buffer 0
    tdm_load_2d(0u,        A + (size_t)m0b * H_, 32u, 128u, H_);
    tdm_load_2d(16384u,    Wb + (size_t)n0b * H_, 32u, 128u, H_);
  }
#endif

  int cur = 0;
  for (int kc = 0; kc < H_; kc += 32) {
    const bool more = (kc + 32 < H_);
#if defined(HAVE_TDM)
    if (w == 0) wait_tensor0();     // current buffer resident in LDS
#endif
    __syncthreads();                // publish TDM writes; drain prior ds reads
#if defined(HAVE_TDM)
    if (more && w == 0) {           // DMA the next pair under this step's WMMAs
      unsigned nb = (unsigned)(cur ^ 1) * 8192u;
      tdm_load_2d(nb,          A + (size_t)m0b * H_ + kc + 32, 32u, 128u, H_);
      tdm_load_2d(16384u + nb, Wb + (size_t)n0b * H_ + kc + 32, 32u, 128u, H_);
    }
#else
    {  // fallback: cooperative copy of the current tile pair
#pragma unroll
      for (int i = 0; i < 2; ++i) {
        int f = tid + 256 * i, r = f >> 2, c = f & 3;
        ((uint4*)(SMEM + cur * 4096))[r * 4 + c] =
            *(const uint4*)(A + (size_t)(m0b + r) * H_ + kc + c * 8);
        ((uint4*)(SMEM + 8192 + cur * 4096))[r * 4 + c] =
            *(const uint4*)(Wb + (size_t)(n0b + r) * H_ + kc + c * 8);
      }
      __syncthreads();
    }
#endif

    const unsigned short* As = SMEM + cur * 4096;
    const unsigned short* Bs = SMEM + 8192 + cur * 4096;
    v16bf a[4], b[2];
#pragma unroll
    for (int i = 0; i < 4; ++i) a[i] = fragA_mem(As + (wm + i * 16) * 32, 32, lane);
#pragma unroll
    for (int j = 0; j < 2; ++j) b[j] = fragB_mem(Bs + (wn + j * 16) * 32, 32, lane);
#pragma unroll
    for (int i = 0; i < 4; ++i)
#pragma unroll
      for (int j = 0; j < 2; ++j) acc[i][j] = wmma_bf16(a[i], b[j], acc[i][j]);

    if (more) cur ^= 1;
  }

  // C/D layout: lanes 0-15 -> N=lane, M=r; lanes 16-31 -> N=lane-16, M=r+8.
#pragma unroll
  for (int i = 0; i < 4; ++i)
#pragma unroll
    for (int j = 0; j < 2; ++j)
#pragma unroll
      for (int r = 0; r < 8; ++r) {
        int m = m0b + wm + i * 16 + r + ((lane & 16) ? 8 : 0);
        int n = n0b + wn + j * 16 + (lane & 15);
        float v = acc[i][j][r];
        if (OUT_MODE == 0) {
          ((float*)Dst)[(size_t)m * H_ + n] = v;
        } else {
          int bb = m >> 11, s = m & (S_ - 1), h = n >> 7, d = n & (HD_ - 1);
          size_t idx = (OUT_MODE == 1)
                           ? (((size_t)(bb * NH_ + h)) * S_ + s) * HD_ + d    // [b,h,s,d]
                           : (((size_t)(bb * NH_ + h)) * HD_ + d) * S_ + s;   // [b,h,d,s]
          ((unsigned short*)Dst)[idx] = f2bf(v);
        }
      }
}

// ---------------------------------------------------------------------------
// RoPE tables. Faithful to source: 'ct' holds sin(freqs), 'st' holds cos(freqs).
// ---------------------------------------------------------------------------
__global__ void rope_tables_kernel(float* __restrict__ ct, float* __restrict__ st) {
  int idx = blockIdx.x * 256 + threadIdx.x;  // S_*64
  int s = idx >> 6, j = idx & 63;
  float inv = __powf(10000.f, -(float)(2 * j) * (1.f / (float)HD_));
  float f = (float)s * inv;
  ct[idx] = __sinf(f);
  st[idx] = __cosf(f);
}

__global__ void rope_apply_kernel(unsigned short* __restrict__ X,
                                  const float* __restrict__ ct,
                                  const float* __restrict__ st) {
  int idx = blockIdx.x * 256 + threadIdx.x;  // B_*NH_*S_*64
  int d = idx & 63;
  int s = (idx >> 6) & (S_ - 1);
  int bh = idx >> 17;                        // / (S_*64)
  size_t base = ((size_t)bh * S_ + s) * HD_;
  float x1 = bf2f(X[base + d]);
  float x2 = bf2f(X[base + d + 64]);
  float c = ct[s * 64 + d], sn = st[s * 64 + d];
  X[base + d]      = f2bf(x1 * c - x2 * sn);
  X[base + d + 64] = f2bf(x1 * sn + x2 * c);
}

// ---------------------------------------------------------------------------
// Flash attention, causal. One wave per (b, h, 32-row Q tile).
// V is pre-transposed ([b,h,d,s]) so BOTH matmuls consume B-fragments as two
// contiguous global_load_b128 per fragment. PV j-loop is software-pipelined
// (prefetch vf(j+1) under the WMMAs of j); softmax reductions are step-outer /
// row-inner so 8 independent shuffles issue per wait.
// ---------------------------------------------------------------------------
__global__ __launch_bounds__(32) void flash_attn_kernel(
    const unsigned short* __restrict__ Qh, const unsigned short* __restrict__ Kh,
    const unsigned short* __restrict__ Vt, unsigned short* __restrict__ AO) {
  __shared__ __align__(16) unsigned short Ps[32 * 32];
  const int lane = threadIdx.x;
  const int bh = blockIdx.x >> 6;        // / (S_/32)
  const int m0 = (blockIdx.x & 63) * 32;
  const unsigned short* Qp = Qh + (size_t)bh * S_ * HD_;
  const unsigned short* Kp = Kh + (size_t)bh * S_ * HD_;
  const unsigned short* Vp = Vt + (size_t)bh * (size_t)HD_ * S_;  // [d][s]
  const float scale = 0.088388347648318447f;  // 128^-0.5

  v16bf qf[2][4];
#pragma unroll
  for (int mt = 0; mt < 2; ++mt)
#pragma unroll
    for (int dc = 0; dc < 4; ++dc)
      qf[mt][dc] = fragA_mem(Qp + (size_t)(m0 + mt * 16) * HD_ + dc * 32, HD_, lane);

  float mrow[2][8], lrow[2][8];
  v8f o[2][8];
#pragma unroll
  for (int mt = 0; mt < 2; ++mt) {
#pragma unroll
    for (int r = 0; r < 8; ++r) { mrow[mt][r] = -INFINITY; lrow[mt][r] = 0.f; }
#pragma unroll
    for (int j = 0; j < 8; ++j) o[mt][j] = v8f_zero();
  }

  for (int kc = 0; kc <= m0; kc += 32) {
    // All K B-fragments of the chunk up front -> loads issue as one clause.
    v16bf kf[2][4];
#pragma unroll
    for (int nt = 0; nt < 2; ++nt)
#pragma unroll
      for (int dc = 0; dc < 4; ++dc)
        kf[nt][dc] =
            fragB_mem(Kp + (size_t)(kc + nt * 16) * HD_ + dc * 32, HD_, lane);

    // scores (32 x 32) = Q * K^T over hd=128
    v8f sc[2][2];
#pragma unroll
    for (int mt = 0; mt < 2; ++mt)
#pragma unroll
      for (int nt = 0; nt < 2; ++nt) {
        v8f s4 = v8f_zero();
#pragma unroll
        for (int dc = 0; dc < 4; ++dc) s4 = wmma_bf16(qf[mt][dc], kf[nt][dc], s4);
        sc[mt][nt] = s4;
      }

    // scale; causal mask only on the diagonal chunk
    if (kc == m0) {
#pragma unroll
      for (int mt = 0; mt < 2; ++mt)
#pragma unroll
        for (int nt = 0; nt < 2; ++nt)
#pragma unroll
          for (int r = 0; r < 8; ++r) {
            int jg = kc + nt * 16 + (lane & 15);
            int mg = m0 + mt * 16 + r + ((lane & 16) ? 8 : 0);
            sc[mt][nt][r] = (jg <= mg) ? sc[mt][nt][r] * scale : -INFINITY;
          }
    } else {
#pragma unroll
      for (int mt = 0; mt < 2; ++mt)
#pragma unroll
        for (int nt = 0; nt < 2; ++nt)
#pragma unroll
          for (int r = 0; r < 8; ++r) sc[mt][nt][r] *= scale;
    }

    // online softmax; reductions are xor-step OUTER so 8 independent
    // ds-shuffles issue back-to-back per step (one wait instead of eight).
#pragma unroll
    for (int mt = 0; mt < 2; ++mt) {
      float mx[8], mnew[8], alpha[8];
#pragma unroll
      for (int r = 0; r < 8; ++r) mx[r] = fmaxf(sc[mt][0][r], sc[mt][1][r]);
#pragma unroll
      for (int off = 8; off > 0; off >>= 1) {
        float t[8];
#pragma unroll
        for (int r = 0; r < 8; ++r) t[r] = __shfl_xor(mx[r], off, 32);
#pragma unroll
        for (int r = 0; r < 8; ++r) mx[r] = fmaxf(mx[r], t[r]);
      }
#pragma unroll
      for (int r = 0; r < 8; ++r) {
        mnew[r] = fmaxf(mrow[mt][r], mx[r]);
        alpha[r] = __expf(mrow[mt][r] - mnew[r]);
        mrow[mt][r] = mnew[r];
      }
#pragma unroll
      for (int nt = 0; nt < 2; ++nt)
#pragma unroll
        for (int r = 0; r < 8; ++r) {
          float p = __expf(sc[mt][nt][r] - mnew[r]);
          sc[mt][nt][r] = p;
          Ps[(mt * 16 + r + ((lane & 16) ? 8 : 0)) * 32 + nt * 16 + (lane & 15)] =
              f2bf(p);
        }
      float sum[8];
#pragma unroll
      for (int r = 0; r < 8; ++r) sum[r] = sc[mt][0][r] + sc[mt][1][r];
#pragma unroll
      for (int off = 8; off > 0; off >>= 1) {
        float t[8];
#pragma unroll
        for (int r = 0; r < 8; ++r) t[r] = __shfl_xor(sum[r], off, 32);
#pragma unroll
        for (int r = 0; r < 8; ++r) sum[r] += t[r];
      }
#pragma unroll
      for (int r = 0; r < 8; ++r) {
        lrow[mt][r] = lrow[mt][r] * alpha[r] + sum[r];
#pragma unroll
        for (int j = 0; j < 8; ++j) o[mt][j][r] *= alpha[r];
      }
    }
    __syncthreads();  // order Ps stores -> loads

    // O += P(32x32) * V(32x128); V fragments pipelined one tile ahead.
    v16bf pa[2];
#pragma unroll
    for (int mt = 0; mt < 2; ++mt) pa[mt] = fragA_mem(Ps + mt * 16 * 32, 32, lane);
    v16bf vf = fragB_mem(Vp + 0 * S_ + kc, S_, lane);
#pragma unroll
    for (int j = 0; j < 8; ++j) {
      v16bf vfn;
      if (j < 7) vfn = fragB_mem(Vp + (size_t)((j + 1) * 16) * S_ + kc, S_, lane);
      o[0][j] = wmma_bf16(pa[0], vf, o[0][j]);
      o[1][j] = wmma_bf16(pa[1], vf, o[1][j]);
      if (j < 7) vf = vfn;
    }
    __syncthreads();  // Ps reads drained before next chunk rewrites it
  }

  const int b = bh >> 4, h = bh & 15;
#pragma unroll
  for (int mt = 0; mt < 2; ++mt)
#pragma unroll
    for (int j = 0; j < 8; ++j)
#pragma unroll
      for (int r = 0; r < 8; ++r) {
        int srow = m0 + mt * 16 + r + ((lane & 16) ? 8 : 0);
        int col = h * HD_ + j * 16 + (lane & 15);
        AO[((size_t)(b * S_ + srow)) * H_ + col] = f2bf(o[mt][j][r] / lrow[mt][r]);
      }
}

// ---------------------------------------------------------------------------
extern "C" void kernel_launch(void* const* d_in, const int* in_sizes, int n_in,
                              void* d_out, int out_size, void* d_ws, size_t ws_size,
                              hipStream_t stream) {
  (void)in_sizes; (void)n_in; (void)out_size; (void)ws_size;
  const float* x  = (const float*)d_in[0];
  const float* Wq = (const float*)d_in[1];
  const float* Wk = (const float*)d_in[2];
  const float* Wv = (const float*)d_in[3];
  const float* Wo = (const float*)d_in[4];

  // Workspace carve-up (bf16 as ushort):
  //  Qb,Kb,Vt,AO: 4 x 32MiB;  xb: 32MiB;  4 weights: 4 x 8MiB;  tables: 1MiB.
  const size_t QKV = (size_t)B_ * NH_ * S_ * HD_;   // == BS_*H_
  const size_t WSZ = (size_t)H_ * H_;
  unsigned short* Qb  = (unsigned short*)d_ws;
  unsigned short* Kb  = Qb + QKV;
  unsigned short* Vt  = Kb + QKV;
  unsigned short* AO  = Vt + QKV;
  unsigned short* xb  = AO + QKV;
  unsigned short* Wqb = xb + QKV;
  unsigned short* Wkb = Wqb + WSZ;
  unsigned short* Wvb = Wkb + WSZ;
  unsigned short* Wob = Wvb + WSZ;
  float* ct = (float*)(Wob + WSZ);
  float* st = ct + (size_t)S_ * (HD_ / 2);

  // bf16 conversions (one-time; ~192MiB of traffic)
  cvt_bf16_kernel<<<(int)(QKV / 4 / 256), 256, 0, stream>>>(x, xb);
  cvt_bf16_kernel<<<(int)(WSZ / 4 / 256), 256, 0, stream>>>(Wq, Wqb);
  cvt_bf16_kernel<<<(int)(WSZ / 4 / 256), 256, 0, stream>>>(Wk, Wkb);
  cvt_bf16_kernel<<<(int)(WSZ / 4 / 256), 256, 0, stream>>>(Wv, Wvb);
  cvt_bf16_kernel<<<(int)(WSZ / 4 / 256), 256, 0, stream>>>(Wo, Wob);

  rope_tables_kernel<<<(S_ * (HD_ / 2)) / 256, 256, 0, stream>>>(ct, st);

  dim3 gg(BS_ / 128, H_ / 128, 1);
  gemm_bf16<1><<<gg, 256, 0, stream>>>(xb, Wqb, Qb);   // Q: [b,h,s,d]
  gemm_bf16<1><<<gg, 256, 0, stream>>>(xb, Wkb, Kb);   // K: [b,h,s,d]
  gemm_bf16<2><<<gg, 256, 0, stream>>>(xb, Wvb, Vt);   // V: [b,h,d,s] (transposed)

  const int rope_blocks = (B_ * NH_ * S_ * (HD_ / 2)) / 256;
  rope_apply_kernel<<<rope_blocks, 256, 0, stream>>>(Qb, ct, st);
  rope_apply_kernel<<<rope_blocks, 256, 0, stream>>>(Kb, ct, st);

  flash_attn_kernel<<<B_ * NH_ * (S_ / 32), 32, 0, stream>>>(Qb, Kb, Vt, AO);

  gemm_bf16<0><<<gg, 256, 0, stream>>>(AO, Wob, (float*)d_out);
}
